// WindowAttention_41120016892061
// MI455X (gfx1250) — matrix operand order repaired and checked by reading, not compiled, not run on hardware
//
#include <hip/hip_runtime.h>
#include <math.h>

// ---------------------------------------------------------------------------
// Types for CDNA5 WMMA (wave32, v_wmma_f32_16x16x32_bf16)
// ---------------------------------------------------------------------------
typedef __bf16  v16bf __attribute__((ext_vector_type(16)));
typedef __bf16  v8bf  __attribute__((ext_vector_type(8)));
typedef __bf16  v2bf  __attribute__((ext_vector_type(2)));
typedef float   v8f   __attribute__((ext_vector_type(8)));
typedef float   v4f   __attribute__((ext_vector_type(4)));

#define NB_C   384
#define NB_3C  1152
#define NB_HW  4096
#define NB_B   16
#define NB_NH  12
#define NB_D   32

#define BM 128
#define BN 128
#define KC 32
#define LSTR 40   // padded LDS row stride (halves); 80B, multiple of 16B

__device__ __forceinline__ v8f wmma_bf16(v16bf a, v16bf b, v8f c) {
    return __builtin_amdgcn_wmma_f32_16x16x32_bf16(
        false, a, false, b, (short)0, c, false, false);
}

// Low 32 bits of a flat shared-aperture address == DS-relative LDS offset.
__device__ __forceinline__ unsigned lds_off(const void* p) {
    return (unsigned)(size_t)p;
}

// CDNA5 async copy: 16B global -> LDS, tracked by ASYNCcnt (bypasses VGPRs).
__device__ __forceinline__ void async_ld_b128(unsigned lds, const void* gsrc) {
    asm volatile("global_load_async_to_lds_b128 %0, %1, off"
                 :: "v"(lds), "v"(gsrc) : "memory");
}
__device__ __forceinline__ void wait_async0() {
    asm volatile("s_wait_asynccnt 0x0" ::: "memory");
}

// A fragment (16x32 bf16): lane l holds row m=l&15; K offset (l>>4)*8.
__device__ __forceinline__ v16bf load_frag_A(const __bf16* base, int rs, int lane) {
    int m = lane & 15;
    int koff = (lane >> 4) * 8;
    const __bf16* p = base + m * rs + koff;
    union { v16bf v; v8bf h[2]; } u;
    u.h[0] = *(const v8bf*)(p);
    u.h[1] = *(const v8bf*)(p + 16);
    return u.v;
}

// B fragment (32x16 bf16): lane l holds col n=l&15; K offset (l>>4)*8,
// halves K koff..koff+7 and koff+16..koff+23.
__device__ __forceinline__ v16bf load_frag_B(const __bf16* base, int rs, int lane) {
    int n = lane & 15;
    int k2 = (lane >> 4) * 8;
    const __bf16* p = base + n * rs + k2;
    union { v16bf v; v8bf h[2]; } u;
    u.h[0] = *(const v8bf*)(p);
    u.h[1] = *(const v8bf*)(p + 16);
    return u.v;
}

// ---------------------------------------------------------------------------
// Kernel 0: convert fp32 weights to bf16
// ---------------------------------------------------------------------------
__global__ void cvt_weights_kernel(const float* __restrict__ wq,
                                   const float* __restrict__ wp,
                                   __bf16* __restrict__ wqb,
                                   __bf16* __restrict__ wpb) {
    int i = blockIdx.x * 256 + threadIdx.x;
    if (i < NB_3C * NB_C) wqb[i] = (__bf16)wq[i];
    if (i < NB_C * NB_C)  wpb[i] = (__bf16)wp[i];
}

// ---------------------------------------------------------------------------
// Kernel 1: QKV GEMM.  qkvT[b][p][o] = sum_c w_qkv[o][c] * x[b][c][p] + bias[o]
//   Double-buffered K loop: A tile async-to-LDS; B tile (f32->bf16 transpose)
//   software-pipelined through VGPRs.
// ---------------------------------------------------------------------------
__global__ __launch_bounds__(256)
void qkv_gemm_kernel(const float* __restrict__ x,
                     const __bf16* __restrict__ wb,
                     const float* __restrict__ bias,
                     __bf16* __restrict__ qkvT) {
    __shared__ __bf16 As[2][BM * LSTR];
    __shared__ __bf16 Bs[2][BN * LSTR];

    const int tid  = threadIdx.x;
    const int lane = tid & 31;
    const int wid  = tid >> 5;
    const int wm   = wid >> 2;
    const int wn   = wid & 3;
    const int mbase = blockIdx.x * BM;
    const int pbase = blockIdx.y * BN;
    const int b     = blockIdx.z;

    const float* xb = x + (size_t)b * NB_C * NB_HW;

    // staging coordinates
    const int am  = tid >> 1;            // A row within tile
    const int akh = (tid & 1) * 16;      // A k-half
    const int bk  = (tid >> 4) * 2;      // B k pair (even)
    const int bn0 = (tid & 15) * 8;      // B n group of 8
    const __bf16* aSrc = wb + (size_t)(mbase + am) * NB_C + akh;
    const float*  bSrc = xb + (size_t)bk * NB_HW + pbase + bn0;

    v4f rg[4];   // pipelined B registers (2 k-rows x 8 n)

    v8f acc[4][2];
#pragma unroll
    for (int i = 0; i < 4; ++i)
#pragma unroll
        for (int j = 0; j < 2; ++j) acc[i][j] = 0.f;

    // ---- prologue: chunk 0 ----
    {
        unsigned l0 = lds_off(&As[0][am * LSTR + akh]);
        async_ld_b128(l0,      aSrc);
        async_ld_b128(l0 + 16, aSrc + 8);
        const float* r0 = bSrc;
        rg[0] = *(const v4f*)(r0);
        rg[1] = *(const v4f*)(r0 + 4);
        rg[2] = *(const v4f*)(r0 + NB_HW);
        rg[3] = *(const v4f*)(r0 + NB_HW + 4);
#pragma unroll
        for (int i = 0; i < 8; ++i) {
            v2bf pk;
            pk.x = (__bf16)((i < 4) ? rg[0][i] : rg[1][i - 4]);
            pk.y = (__bf16)((i < 4) ? rg[2][i] : rg[3][i - 4]);
            *(v2bf*)&Bs[0][(bn0 + i) * LSTR + bk] = pk;
        }
    }

    for (int kc = 0, it = 0; kc < NB_C; kc += KC, ++it) {
        const int buf = it & 1;
        wait_async0();
        __syncthreads();

        const bool next = (kc + KC) < NB_C;
        if (next) {
            // async A for next chunk into other buffer
            const __bf16* srcA = aSrc + kc + KC;
            unsigned l0 = lds_off(&As[buf ^ 1][am * LSTR + akh]);
            async_ld_b128(l0,      srcA);
            async_ld_b128(l0 + 16, srcA + 8);
            // global f32 loads for next B chunk (consumed after compute)
            const float* r0 = bSrc + (size_t)(kc + KC) * NB_HW;
            rg[0] = *(const v4f*)(r0);
            rg[1] = *(const v4f*)(r0 + 4);
            rg[2] = *(const v4f*)(r0 + NB_HW);
            rg[3] = *(const v4f*)(r0 + NB_HW + 4);
        }

        v16bf afr[4], bfr[2];
#pragma unroll
        for (int i = 0; i < 4; ++i)
            afr[i] = load_frag_A(&As[buf][(wm * 64 + i * 16) * LSTR], LSTR, lane);
#pragma unroll
        for (int j = 0; j < 2; ++j)
            bfr[j] = load_frag_B(&Bs[buf][(wn * 32 + j * 16) * LSTR], LSTR, lane);
#pragma unroll
        for (int i = 0; i < 4; ++i)
#pragma unroll
            for (int j = 0; j < 2; ++j)
                acc[i][j] = wmma_bf16(afr[i], bfr[j], acc[i][j]);

        if (next) {
#pragma unroll
            for (int i = 0; i < 8; ++i) {
                v2bf pk;
                pk.x = (__bf16)((i < 4) ? rg[0][i] : rg[1][i - 4]);
                pk.y = (__bf16)((i < 4) ? rg[2][i] : rg[3][i - 4]);
                *(v2bf*)&Bs[buf ^ 1][(bn0 + i) * LSTR + bk] = pk;
            }
        }
        __syncthreads();
    }

    // ---- epilogue: bias add + bf16, transposed store [p][o] ----
    const int r0 = (lane >> 4) * 8;
    const int nn = lane & 15;
#pragma unroll
    for (int i = 0; i < 4; ++i) {
        int obase = mbase + wm * 64 + i * 16 + r0;
        v4f bi0 = *(const v4f*)&bias[obase];
        v4f bi1 = *(const v4f*)&bias[obase + 4];
#pragma unroll
        for (int j = 0; j < 2; ++j) {
            int p = pbase + wn * 32 + j * 16 + nn;
            v8bf outp;
#pragma unroll
            for (int r = 0; r < 4; ++r) outp[r] = (__bf16)(acc[i][j][r] + bi0[r]);
#pragma unroll
            for (int r = 4; r < 8; ++r) outp[r] = (__bf16)(acc[i][j][r] + bi1[r - 4]);
            *(v8bf*)(qkvT + ((size_t)b * NB_HW + p) * NB_3C + obase) = outp;
        }
    }
}

// ---------------------------------------------------------------------------
// Kernel 2: per-window attention. 1 wave per (b, head, window).
//   Q/K staged with async-to-LDS; V transposed through VGPRs.
// ---------------------------------------------------------------------------
#define SPAD 68

__global__ __launch_bounds__(32)
void win_attn_kernel(const __bf16* __restrict__ qkvT, __bf16* __restrict__ oT) {
    __shared__ __bf16 Qs[64 * LSTR];
    __shared__ __bf16 Ks[64 * LSTR];
    __shared__ __bf16 Vt[32 * 72];
    __shared__ __bf16 Ps[64 * 72];
    __shared__ float  Sf[64 * SPAD];
    __shared__ float  Sinv[64];

    const int lane = threadIdx.x;
    const int w    = blockIdx.x;
    const int hi   = w >> 3, wi = w & 7;
    const int head = blockIdx.y;
    const int b    = blockIdx.z;

    const __bf16* qb = qkvT + (size_t)b * NB_HW * NB_3C;
    const int qc = head * NB_D;
    const int kc = NB_C + head * NB_D;
    const int vc = 2 * NB_C + head * NB_D;

    // ---- stage: async Q,K ([token][d]); V transposed via VGPRs ----
#pragma unroll
    for (int tt = 0; tt < 2; ++tt) {
        int t  = lane + tt * 32;
        int sh = t >> 3, sw = t & 7;
        int p  = (sh * 8 + hi) * 64 + (sw * 8 + wi);
        const __bf16* rowp = qb + (size_t)p * NB_3C;
#pragma unroll
        for (int c = 0; c < 4; ++c) {
            async_ld_b128(lds_off(&Qs[t * LSTR + c * 8]), rowp + qc + c * 8);
            async_ld_b128(lds_off(&Ks[t * LSTR + c * 8]), rowp + kc + c * 8);
        }
        v8bf vv[4];
#pragma unroll
        for (int c = 0; c < 4; ++c) vv[c] = *(const v8bf*)(rowp + vc + c * 8);
#pragma unroll
        for (int d = 0; d < 32; ++d) Vt[d * 72 + t] = vv[d >> 3][d & 7];
    }
    wait_async0();
    __syncthreads();

    // ---- S = Q K^T ----
    v16bf aq[4], bk[4];
#pragma unroll
    for (int i = 0; i < 4; ++i) aq[i] = load_frag_A(&Qs[i * 16 * LSTR], LSTR, lane);
#pragma unroll
    for (int j = 0; j < 4; ++j) bk[j] = load_frag_B(&Ks[j * 16 * LSTR], LSTR, lane);

    const int r0 = (lane >> 4) * 8;
    const int nn = lane & 15;
#pragma unroll
    for (int i = 0; i < 4; ++i) {
#pragma unroll
        for (int j = 0; j < 4; ++j) {
            v8f s = 0.f;
            s = wmma_bf16(aq[i], bk[j], s);
#pragma unroll
            for (int r = 0; r < 8; ++r)
                Sf[(i * 16 + r0 + r) * SPAD + j * 16 + nn] = s[r];
        }
    }
    __syncthreads();

    // ---- softmax (2 rows per lane); defer 1/sum to PV epilogue ----
    const float cl2 = 0.17677669529663687f * 1.4426950408889634f;
#pragma unroll
    for (int rr = 0; rr < 2; ++rr) {
        int q = lane * 2 + rr;
        const float* row = &Sf[q * SPAD];
        float mx = -3.4e38f;
#pragma unroll
        for (int k = 0; k < 64; ++k) mx = fmaxf(mx, row[k]);
        float sum = 0.f;
#pragma unroll
        for (int k = 0; k < 64; k += 2) {
            float e0 = exp2f((row[k] - mx) * cl2);
            float e1 = exp2f((row[k + 1] - mx) * cl2);
            sum += e0 + e1;
            v2bf pk;
            pk.x = (__bf16)e0;
            pk.y = (__bf16)e1;
            *(v2bf*)&Ps[q * 72 + k] = pk;
        }
        Sinv[q] = 1.0f / sum;
    }
    __syncthreads();

    // ---- O = P V ----
    v8f o[4][2];
#pragma unroll
    for (int i = 0; i < 4; ++i)
#pragma unroll
        for (int j = 0; j < 2; ++j) o[i][j] = 0.f;
#pragma unroll
    for (int kk = 0; kk < 2; ++kk) {
        v16bf ap[4], bv[2];
#pragma unroll
        for (int i = 0; i < 4; ++i)
            ap[i] = load_frag_A(&Ps[i * 16 * 72 + kk * 32], 72, lane);
#pragma unroll
        for (int j = 0; j < 2; ++j)
            bv[j] = load_frag_B(&Vt[j * 16 * 72 + kk * 32], 72, lane);
#pragma unroll
        for (int i = 0; i < 4; ++i)
#pragma unroll
            for (int j = 0; j < 2; ++j)
                o[i][j] = wmma_bf16(ap[i], bv[j], o[i][j]);
    }

    // ---- epilogue: normalize, scatter to blocked output pixel ----
#pragma unroll
    for (int i = 0; i < 4; ++i) {
#pragma unroll
        for (int j = 0; j < 2; ++j) {
#pragma unroll
            for (int r = 0; r < 8; ++r) {
                int m  = i * 16 + r0 + r;
                int sh = m >> 3, sw = m & 7;
                int pp = (hi * 8 + sh) * 64 + (wi * 8 + sw);
                float val = o[i][j][r] * Sinv[m];
                oT[((size_t)b * NB_HW + pp) * NB_C + head * NB_D + j * 16 + nn] =
                    (__bf16)val;
            }
        }
    }
}

// ---------------------------------------------------------------------------
// Kernel 3: projection GEMM, fully async-staged + double buffered.
//   out[b][o][p] = sum_c w_proj[o][c] * oT[b][p][c] + bias[o]
// ---------------------------------------------------------------------------
__global__ __launch_bounds__(256)
void proj_gemm_kernel(const __bf16* __restrict__ oT,
                      const __bf16* __restrict__ wb,
                      const float* __restrict__ bias,
                      float* __restrict__ out) {
    __shared__ __bf16 As[2][BM * LSTR];
    __shared__ __bf16 Bs[2][BN * LSTR];

    const int tid  = threadIdx.x;
    const int lane = tid & 31;
    const int wid  = tid >> 5;
    const int wm   = wid >> 2;
    const int wn   = wid & 3;
    const int mbase = blockIdx.x * BM;
    const int pbase = blockIdx.y * BN;
    const int b     = blockIdx.z;

    const int m  = tid >> 1;
    const int kh = (tid & 1) * 16;
    const __bf16* aSrc = wb + (size_t)(mbase + m) * NB_C + kh;
    const __bf16* bSrc = oT + ((size_t)b * NB_HW + pbase + m) * NB_C + kh;

    v8f acc[4][2];
#pragma unroll
    for (int i = 0; i < 4; ++i)
#pragma unroll
        for (int j = 0; j < 2; ++j) acc[i][j] = 0.f;

    // prologue: async both tiles of chunk 0
    {
        unsigned la = lds_off(&As[0][m * LSTR + kh]);
        unsigned lb = lds_off(&Bs[0][m * LSTR + kh]);
        async_ld_b128(la,      aSrc);
        async_ld_b128(la + 16, aSrc + 8);
        async_ld_b128(lb,      bSrc);
        async_ld_b128(lb + 16, bSrc + 8);
    }

    for (int kcc = 0, it = 0; kcc < NB_C; kcc += KC, ++it) {
        const int buf = it & 1;
        wait_async0();
        __syncthreads();

        if ((kcc + KC) < NB_C) {
            unsigned la = lds_off(&As[buf ^ 1][m * LSTR + kh]);
            unsigned lb = lds_off(&Bs[buf ^ 1][m * LSTR + kh]);
            async_ld_b128(la,      aSrc + kcc + KC);
            async_ld_b128(la + 16, aSrc + kcc + KC + 8);
            async_ld_b128(lb,      bSrc + kcc + KC);
            async_ld_b128(lb + 16, bSrc + kcc + KC + 8);
        }

        v16bf afr[4], bfr[2];
#pragma unroll
        for (int i = 0; i < 4; ++i)
            afr[i] = load_frag_A(&As[buf][(wm * 64 + i * 16) * LSTR], LSTR, lane);
#pragma unroll
        for (int j = 0; j < 2; ++j)
            bfr[j] = load_frag_B(&Bs[buf][(wn * 32 + j * 16) * LSTR], LSTR, lane);
#pragma unroll
        for (int i = 0; i < 4; ++i)
#pragma unroll
            for (int j = 0; j < 2; ++j)
                acc[i][j] = wmma_bf16(afr[i], bfr[j], acc[i][j]);
        __syncthreads();
    }

    // epilogue: fp32 NCHW output with bias
    const int r0 = (lane >> 4) * 8;
    const int nn = lane & 15;
#pragma unroll
    for (int i = 0; i < 4; ++i) {
        int obase = mbase + wm * 64 + i * 16 + r0;
        v4f bi0 = *(const v4f*)&bias[obase];
        v4f bi1 = *(const v4f*)&bias[obase + 4];
#pragma unroll
        for (int j = 0; j < 2; ++j) {
            int p = pbase + wn * 32 + j * 16 + nn;
#pragma unroll
            for (int r = 0; r < 8; ++r) {
                float bv = (r < 4) ? bi0[r] : bi1[r - 4];
                out[((size_t)b * NB_C + obase + r) * NB_HW + p] = acc[i][j][r] + bv;
            }
        }
    }
}

// ---------------------------------------------------------------------------
// Launch
// ---------------------------------------------------------------------------
extern "C" void kernel_launch(void* const* d_in, const int* in_sizes, int n_in,
                              void* d_out, int out_size, void* d_ws, size_t ws_size,
                              hipStream_t stream) {
    const float* x      = (const float*)d_in[0];
    const float* w_qkv  = (const float*)d_in[1];
    const float* b_qkv  = (const float*)d_in[2];
    const float* w_proj = (const float*)d_in[3];
    const float* b_proj = (const float*)d_in[4];
    float* out = (float*)d_out;

    char* ws = (char*)d_ws;
    __bf16* wqb  = (__bf16*)(ws);                              //  884736 B
    __bf16* wpb  = (__bf16*)(ws + 884736);                     //  294912 B
    __bf16* qkvT = (__bf16*)(ws + 1179648);                    // 150994944 B
    __bf16* oTs  = (__bf16*)(ws + 1179648 + 150994944ull);     //  50331648 B

    cvt_weights_kernel<<<(NB_3C * NB_C + 255) / 256, 256, 0, stream>>>(
        w_qkv, w_proj, wqb, wpb);

    dim3 g1(NB_3C / BM, NB_HW / BN, NB_B);
    qkv_gemm_kernel<<<g1, 256, 0, stream>>>(x, wqb, b_qkv, qkvT);

    dim3 g2(64, NB_NH, NB_B);
    win_attn_kernel<<<g2, 32, 0, stream>>>(qkvT, oTs);

    dim3 g3(NB_C / BM, NB_HW / BN, NB_B);
    proj_gemm_kernel<<<g3, 256, 0, stream>>>(oTs, wpb, b_proj, out);
}
